// HyperSelfAttention_47622597378301
// MI455X (gfx1250) — matrix operand order repaired and checked by reading, not compiled, not run on hardware
//
#include <hip/hip_runtime.h>

typedef __attribute__((ext_vector_type(16))) __bf16 v16bf;
typedef __attribute__((ext_vector_type(8)))  float  v8f;

#define NTOK   4096
#define NPROJ  8
#define BLK    128
#define SAMP   128
#define HD     64
#define SCALE  0.125f
#define NEG_BIG (-3.4028235e38f)
#define LOG_N_OVER_M 3.4657359028f   // log(4096/128)

// ---------------------------------------------------------------------------
// Kernel 1: angular LSH hash (Gray-coded 8-bit bucket) for Q and K rows.
// ---------------------------------------------------------------------------
__global__ __launch_bounds__(256) void hash_kernel(const float* __restrict__ qkv,
                                                   const float* __restrict__ proj,
                                                   int* __restrict__ hq,
                                                   int* __restrict__ hk) {
  int tid  = blockIdx.x * 256 + threadIdx.x;     // 0 .. 262143
  int comp = tid >> 17;                          // 0 = q, 1 = k
  int r    = tid & ((1 << 17) - 1);
  int bh   = r >> 12;
  int n    = r & (NTOK - 1);
  int b    = bh >> 4, h = bh & 15;
  const float* x = qkv + (((size_t)b * NTOK + n) * 3 + comp) * 1024 + (size_t)h * 64;
  float acc[NPROJ];
#pragma unroll
  for (int p = 0; p < NPROJ; ++p) acc[p] = 0.f;
#pragma unroll 8
  for (int d = 0; d < HD; ++d) {
    float xv = x[d];
#pragma unroll
    for (int p = 0; p < NPROJ; ++p) acc[p] += xv * proj[d * NPROJ + p];
  }
  int e = 0;
#pragma unroll
  for (int p = 0; p < NPROJ; ++p) e |= (acc[p] > 0.f) ? (1 << p) : 0;
  int hsh = e ^ (e >> 1);                        // binary-reflected Gray code
  (comp ? hk : hq)[bh * NTOK + n] = hsh;
}

// ---------------------------------------------------------------------------
// Kernel 2: stable counting/radix sort of 4096 8-bit keys per (b,h) x {q,k}.
// Key packed as (hash<<16)|pos so stability matches jnp.argsort(stable=True).
// ---------------------------------------------------------------------------
__global__ __launch_bounds__(256) void sort_kernel(const int* __restrict__ hq,
                                                   const int* __restrict__ hk,
                                                   int* __restrict__ qidx,
                                                   int* __restrict__ kidx) {
  __shared__ unsigned bufA[NTOK];
  __shared__ unsigned bufB[NTOK];
  __shared__ unsigned hist[16 * 256];
  __shared__ unsigned aux[256];
  int bh = blockIdx.x;
  const int* hsrc = blockIdx.y ? hk : hq;
  int* outIdx     = blockIdx.y ? kidx : qidx;
  int t = threadIdx.x;
#pragma unroll
  for (int i = 0; i < 16; ++i) {
    int n = t * 16 + i;
    bufA[n] = ((unsigned)hsrc[bh * NTOK + n] << 16) | (unsigned)n;
  }
  __syncthreads();
  unsigned* src = bufA;
  unsigned* dst = bufB;
  for (int pass = 0; pass < 2; ++pass) {
    int shift = 16 + 4 * pass;
    unsigned keys[16], cnt[16];
#pragma unroll
    for (int d = 0; d < 16; ++d) cnt[d] = 0;
#pragma unroll
    for (int i = 0; i < 16; ++i) {
      keys[i] = src[t * 16 + i];
      cnt[(keys[i] >> shift) & 15]++;
    }
#pragma unroll
    for (int d = 0; d < 16; ++d) hist[d * 256 + t] = cnt[d];
    __syncthreads();
    // block-wide exclusive scan of the 4096-entry flattened (digit-major) hist
    unsigned loc[16], run = 0;
#pragma unroll
    for (int i = 0; i < 16; ++i) { unsigned v = hist[t * 16 + i]; loc[i] = run; run += v; }
    aux[t] = run;
    __syncthreads();
    if (t == 0) {
      unsigned s = 0;
      for (int i = 0; i < 256; ++i) { unsigned v = aux[i]; aux[i] = s; s += v; }
    }
    __syncthreads();
    run = aux[t];
#pragma unroll
    for (int i = 0; i < 16; ++i) hist[t * 16 + i] = loc[i] + run;
    __syncthreads();
    unsigned base[16];
#pragma unroll
    for (int d = 0; d < 16; ++d) base[d] = hist[d * 256 + t];
#pragma unroll
    for (int i = 0; i < 16; ++i) {
      unsigned kk = keys[i];
      unsigned d  = (kk >> shift) & 15;
      dst[base[d]++] = kk;              // chunk-sequential -> stable
    }
    __syncthreads();
    unsigned* tmp = src; src = dst; dst = tmp;
  }
#pragma unroll
  for (int i = 0; i < 16; ++i) {
    int n = t * 16 + i;
    outIdx[bh * NTOK + n] = (int)(src[n] & 0xFFFFu);
  }
}

// ---------------------------------------------------------------------------
// Kernel 3: fused block-diagonal + sampled-residual attention with WMMA.
// ---------------------------------------------------------------------------
__device__ __forceinline__ float hmax16(float v) {
  v = fmaxf(v, __shfl_xor(v, 1, 32));
  v = fmaxf(v, __shfl_xor(v, 2, 32));
  v = fmaxf(v, __shfl_xor(v, 4, 32));
  v = fmaxf(v, __shfl_xor(v, 8, 32));
  return v;
}
__device__ __forceinline__ float hsum16(float v) {
  v += __shfl_xor(v, 1, 32);
  v += __shfl_xor(v, 2, 32);
  v += __shfl_xor(v, 4, 32);
  v += __shfl_xor(v, 8, 32);
  return v;
}

__global__ __launch_bounds__(256) void attn_kernel(const float* __restrict__ qkv,
                                                   const int* __restrict__ qidx,
                                                   const int* __restrict__ kidx,
                                                   const int* __restrict__ sampled,
                                                   float* __restrict__ out) {
  __shared__ __bf16 pL[8][16][128];    // per-wave private P strips (no barriers)
  const int g  = blockIdx.x;           // sorted block 0..31
  const int bh = blockIdx.y;           // 0..31
  const int b  = bh >> 4, h = bh & 15;
  const int t  = threadIdx.x;
  const int wave = t >> 5;
  const int lane = t & 31;
  const int lo = lane & 15;
  const int hi = lane >> 4;            // which 16-lane half
  const size_t hoff = (size_t)h * 64;

  // ---- A operand: 16 sorted query rows as two 16x32 bf16 tiles (K = d)
  const int qsrow = g * BLK + wave * 16 + lo;
  const int qorig = qidx[bh * NTOK + qsrow];
  const float* qrow = qkv + (((size_t)b * NTOK + qorig) * 3 + 0) * 1024 + hoff;
  v16bf aq0, aq1;
#pragma unroll
  for (int e = 0; e < 16; ++e) {
    int d = e + hi * 16;                      // contiguous k order (matches B)
    aq0[e] = (__bf16)(qrow[d]);
    aq1[e] = (__bf16)(qrow[d + 32]);
  }

  v8f sv[8];
  float lseb[8], sumb[8], lser[8], sumr[8];
  v8f ob[4] = {};
  v8f orr[4] = {};

  // ================= phase 1: block-diagonal attention =================
#pragma unroll
  for (int c = 0; c < 8; ++c) {
    int korig = kidx[bh * NTOK + g * BLK + c * 16 + lo];
    const float* krow = qkv + (((size_t)b * NTOK + korig) * 3 + 1) * 1024 + hoff;
    v16bf b0, b1;
#pragma unroll
    for (int e = 0; e < 16; ++e) {
      int d = e + hi * 16;                    // contiguous k order (matches A)
      b0[e] = (__bf16)(krow[d]);
      b1[e] = (__bf16)(krow[d + 32]);
    }
    v8f acc = {};
    acc = __builtin_amdgcn_wmma_f32_16x16x32_bf16(false, aq0, false, b0, (short)0, acc, false, false);
    acc = __builtin_amdgcn_wmma_f32_16x16x32_bf16(false, aq1, false, b1, (short)0, acc, false, false);
#pragma unroll
    for (int j = 0; j < 8; ++j) acc[j] *= SCALE;
    sv[c] = acc;
  }
#pragma unroll
  for (int j = 0; j < 8; ++j) {
    float m = sv[0][j];
#pragma unroll
    for (int c = 1; c < 8; ++c) m = fmaxf(m, sv[c][j]);
    m = hmax16(m);
    float s = 0.f;
#pragma unroll
    for (int c = 0; c < 8; ++c) { float p = __expf(sv[c][j] - m); sv[c][j] = p; s += p; }
    s = hsum16(s);
    sumb[j] = s;
    lseb[j] = m + __logf(s);
  }
#pragma unroll
  for (int c = 0; c < 8; ++c)
#pragma unroll
    for (int j = 0; j < 8; ++j)
      pL[wave][j + hi * 8][c * 16 + lo] = (__bf16)(sv[c][j]);
  // O_blk = P (16x128) * V (128x64)
#pragma unroll
  for (int kc = 0; kc < 4; ++kc) {
    v16bf ap;
#pragma unroll
    for (int e = 0; e < 16; ++e)
      ap[e] = pL[wave][lo][kc * 32 + e + hi * 16];
    const float* vptr[16];
#pragma unroll
    for (int e = 0; e < 16; ++e) {
      int vorig = kidx[bh * NTOK + g * BLK + kc * 32 + e + hi * 16];
      vptr[e] = qkv + (((size_t)b * NTOK + vorig) * 3 + 2) * 1024 + hoff;
    }
#pragma unroll
    for (int oc = 0; oc < 4; ++oc) {
      v16bf bv;
#pragma unroll
      for (int e = 0; e < 16; ++e) bv[e] = (__bf16)(vptr[e][oc * 16 + lo]);
      ob[oc] = __builtin_amdgcn_wmma_f32_16x16x32_bf16(false, ap, false, bv, (short)0, ob[oc], false, false);
    }
  }

  // ================= phase 2: sampled residual attention =================
#pragma unroll
  for (int c = 0; c < 8; ++c) {
    int spos  = sampled[bh * SAMP + c * 16 + lo];     // position in sorted ks
    int korig = kidx[bh * NTOK + spos];
    const float* krow = qkv + (((size_t)b * NTOK + korig) * 3 + 1) * 1024 + hoff;
    float bias = ((spos >> 7) == g) ? NEG_BIG : 0.f;  // block-collision mask
    v16bf b0, b1;
#pragma unroll
    for (int e = 0; e < 16; ++e) {
      int d = e + hi * 16;
      b0[e] = (__bf16)(krow[d]);
      b1[e] = (__bf16)(krow[d + 32]);
    }
    v8f acc = {};
    acc = __builtin_amdgcn_wmma_f32_16x16x32_bf16(false, aq0, false, b0, (short)0, acc, false, false);
    acc = __builtin_amdgcn_wmma_f32_16x16x32_bf16(false, aq1, false, b1, (short)0, acc, false, false);
#pragma unroll
    for (int j = 0; j < 8; ++j) acc[j] = acc[j] * SCALE + bias;
    sv[c] = acc;
  }
#pragma unroll
  for (int j = 0; j < 8; ++j) {
    float m = sv[0][j];
#pragma unroll
    for (int c = 1; c < 8; ++c) m = fmaxf(m, sv[c][j]);
    m = hmax16(m);
    float s = 0.f;
#pragma unroll
    for (int c = 0; c < 8; ++c) { float p = __expf(sv[c][j] - m); sv[c][j] = p; s += p; }
    s = hsum16(s);
    sumr[j] = s;
    lser[j] = m + __logf(s) + LOG_N_OVER_M;
  }
#pragma unroll
  for (int c = 0; c < 8; ++c)
#pragma unroll
    for (int j = 0; j < 8; ++j)
      pL[wave][j + hi * 8][c * 16 + lo] = (__bf16)(sv[c][j]);
#pragma unroll
  for (int kc = 0; kc < 4; ++kc) {
    v16bf ap;
#pragma unroll
    for (int e = 0; e < 16; ++e)
      ap[e] = pL[wave][lo][kc * 32 + e + hi * 16];
    const float* vptr[16];
#pragma unroll
    for (int e = 0; e < 16; ++e) {
      int spos  = sampled[bh * SAMP + kc * 32 + e + hi * 16];
      int vorig = kidx[bh * NTOK + spos];
      vptr[e] = qkv + (((size_t)b * NTOK + vorig) * 3 + 2) * 1024 + hoff;
    }
#pragma unroll
    for (int oc = 0; oc < 4; ++oc) {
      v16bf bv;
#pragma unroll
      for (int e = 0; e < 16; ++e) bv[e] = (__bf16)(vptr[e][oc * 16 + lo]);
      orr[oc] = __builtin_amdgcn_wmma_f32_16x16x32_bf16(false, ap, false, bv, (short)0, orr[oc], false, false);
    }
  }

  // ================= combine (logaddexp) + un-sort scatter =================
#pragma unroll
  for (int j = 0; j < 8; ++j) {
    float la = lseb[j], lb = lser[j];
    float mx = fmaxf(la, lb);
    float lse = mx + __logf(__expf(la - mx) + __expf(lb - mx));
    float w1 = __expf(la - lse);
    float ca = w1 / sumb[j];              // O tiles hold unnormalized exp sums
    float cb = (1.f - w1) / sumr[j];
    int srow = g * BLK + wave * 16 + j + hi * 8;
    int orig = qidx[bh * NTOK + srow];
    float* orow = out + (((size_t)b * NTOK + orig) * 16 + h) * 64;
#pragma unroll
    for (int oc = 0; oc < 4; ++oc)
      orow[oc * 16 + lo] = ca * ob[oc][j] + cb * orr[oc][j];
  }
}

// ---------------------------------------------------------------------------
extern "C" void kernel_launch(void* const* d_in, const int* in_sizes, int n_in,
                              void* d_out, int out_size, void* d_ws, size_t ws_size,
                              hipStream_t stream) {
  const float* qkv    = (const float*)d_in[0];   // [2,4096,3,16,64] f32
  const float* proj   = (const float*)d_in[1];   // [64,8] f32
  const int*  sampled = (const int*)d_in[2];     // [2,16,128] i32
  float* out = (float*)d_out;                    // [2,4096,16,64] f32

  int* hq   = (int*)d_ws;                        // [32*4096]
  int* hk   = hq + 32 * NTOK;
  int* qidx = hk + 32 * NTOK;
  int* kidx = qidx + 32 * NTOK;                  // total 2 MB scratch

  hash_kernel<<<1024, 256, 0, stream>>>(qkv, proj, hq, hk);
  sort_kernel<<<dim3(32, 2), 256, 0, stream>>>(hq, hk, qidx, kidx);
  attn_kernel<<<dim3(32, 32), 256, 0, stream>>>(qkv, qidx, kidx, sampled, out);
}
